// SSDLayer_44427141710404
// MI455X (gfx1250) — compile-verified, matrix operand order checked by reading, not run on hardware
//
#include <hip/hip_runtime.h>
#include <hip/hip_bf16.h>

// ---------------- problem constants ----------------
#define B_   2
#define L_   4096
#define D_   1024
#define DI_  2048
#define H_   16
#define HD_  128
#define N_   128
#define KCONV 4
#define CS_  256
#define MC_  256
#define NC_  16
#define PD_  8208          // 2*DI + 2*H*N + H
#define ALPHA_ 0.5f

typedef __attribute__((ext_vector_type(16))) __bf16 bf16x16;
typedef __attribute__((ext_vector_type(8)))  float  f32x8;

__device__ __forceinline__ float sigmoidf_(float x) { return 1.0f / (1.0f + __expf(-x)); }
__device__ __forceinline__ float siluf_(float x)    { return x * sigmoidf_(x); }
__device__ __forceinline__ float softplusf_(float x){ return (x > 20.f) ? x : log1pf(__expf(x)); }

// Low 32 bits of a generic pointer to __shared__ == LDS byte offset (ISA 10.2).
__device__ __forceinline__ uint32_t lds_off32(const void* p)
{
    return (uint32_t)(uintptr_t)p;
}

// CDNA5 async DMA: copy 16 bytes per lane from global memory directly into LDS.
// Tracked by ASYNCcnt (CDNA5 ISA).
__device__ __forceinline__ void async_cp16(uint32_t lds_byte_off, const float* gaddr)
{
    asm volatile("global_load_async_to_lds_b128 %0, %1, off"
                 :: "v"(lds_byte_off), "v"(gaddr)
                 : "memory");
}

__device__ __forceinline__ void async_wait0()
{
    asm volatile("s_wait_asynccnt 0" ::: "memory");
}

// ------------------------------------------------------------------
// Generic batched WMMA GEMM:  C[z] (M x N) (+)= rowscale * A[z] @ B[z]
//   A: TRA=false -> [M][K] (lda = row stride), TRA=true -> [K][M] (lda = k stride)
//   B: BKM=false -> [N][K] "NT"  , BKM=true -> [K][N] "NN"
//   C: [M][N] f32, ldc row stride.  beta scales existing C (0 -> overwrite).
//   rsLog: optional per-output-row log-scale, result *= exp(rsLog[z*sRS + row]).
// 128 threads = 4 wave32; block tile 64x64, K-step 32, bf16 WMMA, f32 acc.
// Full tiles are staged with global_load_async_to_lds_b128 (f32), then
// converted in-LDS to bf16; edge tiles fall back to guarded scalar staging.
// ------------------------------------------------------------------
template<bool TRA, bool BKM>
__global__ __launch_bounds__(128)
void gemm_wmma_bf16(const float* __restrict__ A, const float* __restrict__ B,
                    float* __restrict__ C,
                    int M, int N, int K, int lda, int ldb, int ldc,
                    long long sA, long long sB, long long sC,
                    float beta, const float* __restrict__ rsLog, long long sRS)
{
    __shared__ __bf16 As[64][40];
    __shared__ __bf16 Bs[64][40];
    __shared__ __align__(16) float stA[2048];   // 64x32 (or 32x64) f32 staging
    __shared__ __align__(16) float stB[2048];

    const int tid  = threadIdx.x;
    const int lane = tid & 31;
    const int wv   = tid >> 5;       // 0..3
    const int g    = lane >> 4;      // 0/1 lane half
    const int ln16 = lane & 15;
    const int m0 = blockIdx.y * 64;
    const int n0 = blockIdx.x * 64;
    const long long z = blockIdx.z;
    const float* __restrict__ Ab = A + z * sA;
    const float* __restrict__ Bb = B + z * sB;

    f32x8 acc[4] = {};

    for (int k0 = 0; k0 < K; k0 += 32) {
        const bool fullA = (m0 + 64 <= M) && (k0 + 32 <= K);   // uniform
        const bool fullB = (n0 + 64 <= N) && (k0 + 32 <= K);   // uniform

        // ---- stage A ----
        if (fullA) {
#pragma unroll
            for (int i = 0; i < 4; ++i) {
                int chunk = tid + i * 128;            // 0..511, 4 floats each
                if (TRA) {
                    int k = chunk >> 4, mg = (chunk & 15) * 4;
                    async_cp16(lds_off32(&stA[k * 64 + mg]),
                               Ab + (long long)(k0 + k) * lda + (m0 + mg));
                } else {
                    int r = chunk >> 3, cg = (chunk & 7) * 4;
                    async_cp16(lds_off32(&stA[r * 32 + cg]),
                               Ab + (long long)(m0 + r) * lda + (k0 + cg));
                }
            }
        } else {
            for (int e = tid; e < 64 * 32; e += 128) {
                int r = e >> 5, kk = e & 31;
                int gm = m0 + r, gk = k0 + kk;
                float v = 0.f;
                if (gm < M && gk < K)
                    v = TRA ? Ab[(long long)gk * lda + gm] : Ab[(long long)gm * lda + gk];
                As[r][kk] = (__bf16)v;
            }
        }
        // ---- stage B ----
        if (fullB) {
#pragma unroll
            for (int i = 0; i < 4; ++i) {
                int chunk = tid + i * 128;
                if (BKM) {
                    int k = chunk >> 4, ng = (chunk & 15) * 4;
                    async_cp16(lds_off32(&stB[k * 64 + ng]),
                               Bb + (long long)(k0 + k) * ldb + (n0 + ng));
                } else {
                    int r = chunk >> 3, cg = (chunk & 7) * 4;
                    async_cp16(lds_off32(&stB[r * 32 + cg]),
                               Bb + (long long)(n0 + r) * ldb + (k0 + cg));
                }
            }
        } else {
            for (int e = tid; e < 64 * 32; e += 128) {
                int r = e >> 5, kk = e & 31;
                int gn = n0 + r, gk = k0 + kk;
                float v = 0.f;
                if (gn < N && gk < K)
                    v = BKM ? Bb[(long long)gk * ldb + gn] : Bb[(long long)gn * ldb + gk];
                Bs[r][kk] = (__bf16)v;
            }
        }

        // ---- wait for DMA, then in-LDS f32 -> bf16 conversion ----
        if (fullA || fullB) {
            async_wait0();
            __syncthreads();
            if (fullA) {
#pragma unroll
                for (int i = 0; i < 4; ++i) {
                    int chunk = tid + i * 128;
                    if (TRA) {
                        int k = chunk >> 4, mg = (chunk & 15) * 4;
#pragma unroll
                        for (int j = 0; j < 4; ++j)
                            As[mg + j][k] = (__bf16)stA[k * 64 + mg + j];
                    } else {
                        int r = chunk >> 3, cg = (chunk & 7) * 4;
#pragma unroll
                        for (int j = 0; j < 4; ++j)
                            As[r][cg + j] = (__bf16)stA[r * 32 + cg + j];
                    }
                }
            }
            if (fullB) {
#pragma unroll
                for (int i = 0; i < 4; ++i) {
                    int chunk = tid + i * 128;
                    if (BKM) {
                        int k = chunk >> 4, ng = (chunk & 15) * 4;
#pragma unroll
                        for (int j = 0; j < 4; ++j)
                            Bs[ng + j][k] = (__bf16)stB[k * 64 + ng + j];
                    } else {
                        int r = chunk >> 3, cg = (chunk & 7) * 4;
#pragma unroll
                        for (int j = 0; j < 4; ++j)
                            Bs[r][cg + j] = (__bf16)stB[r * 32 + cg + j];
                    }
                }
            }
        }
        __syncthreads();

        // ---- fragments + WMMA (ISA 7.12.2 16-bit 16x32 layout) ----
        bf16x16 af;
        {
            int mloc = wv * 16 + ln16;
#pragma unroll
            for (int v = 0; v < 4; ++v) {
                int kk = g * 8 + 2 * v;
                af[2 * v]     = As[mloc][kk];
                af[2 * v + 1] = As[mloc][kk + 1];
                af[2 * v + 8] = As[mloc][kk + 16];
                af[2 * v + 9] = As[mloc][kk + 17];
            }
        }
#pragma unroll
        for (int j = 0; j < 4; ++j) {
            bf16x16 bf;
            int nloc = j * 16 + ln16;
#pragma unroll
            for (int v = 0; v < 4; ++v) {
                int kk = g * 8 + 2 * v;
                bf[2 * v]     = Bs[nloc][kk];
                bf[2 * v + 1] = Bs[nloc][kk + 1];
                bf[2 * v + 8] = Bs[nloc][kk + 16];
                bf[2 * v + 9] = Bs[nloc][kk + 17];
            }
            acc[j] = __builtin_amdgcn_wmma_f32_16x16x32_bf16(
                false, af, false, bf, (short)0, acc[j], false, false);
        }
        __syncthreads();
    }

    // ---- epilogue: optional per-row exp(log-scale), beta-accumulate ----
#pragma unroll
    for (int j = 0; j < 4; ++j) {
#pragma unroll
        for (int v = 0; v < 8; ++v) {
            int row = m0 + wv * 16 + g * 8 + v;
            int col = n0 + j * 16 + ln16;
            if (row < M && col < N) {
                float r = acc[j][v];
                if (rsLog) r *= __expf(rsLog[z * sRS + row]);
                float* cp = C + z * sC + (long long)row * ldc + col;
                float prev = (beta != 0.f) ? beta * (*cp) : 0.f;
                *cp = prev + r;
            }
        }
    }
}

// ---------------- element-wise / scan helper kernels ----------------

__global__ __launch_bounds__(256)
void rmsnorm_kernel(const float* __restrict__ x, const float* __restrict__ w,
                    float* __restrict__ xn)
{
    __shared__ float red[256];
    long long row = blockIdx.x;                      // b*L + l
    const float* xr = x + row * D_;
    float s = 0.f;
    for (int d = threadIdx.x; d < D_; d += 256) { float v = xr[d]; s += v * v; }
    red[threadIdx.x] = s; __syncthreads();
    for (int o = 128; o > 0; o >>= 1) {
        if (threadIdx.x < o) red[threadIdx.x] += red[threadIdx.x + o];
        __syncthreads();
    }
    float inv = rsqrtf(red[0] / (float)D_ + 1e-6f);
    for (int d = threadIdx.x; d < D_; d += 256)
        xn[row * D_ + d] = xr[d] * inv * w[d];
}

__global__ __launch_bounds__(256)
void conv_silu_kernel(const float* __restrict__ proj, const float* __restrict__ cw,
                      const float* __restrict__ cb, float* __restrict__ vssd)
{
    long long row = blockIdx.x;                      // b*L + l
    int b = (int)(row / L_), l = (int)(row % L_);
    for (int c = threadIdx.x; c < DI_; c += 256) {
        float acc = cb[c];
#pragma unroll
        for (int k = 0; k < KCONV; ++k) {
            int ls = l + k - (KCONV - 1);
            if (ls >= 0)
                acc += cw[c * KCONV + k] * proj[((long long)b * L_ + ls) * PD_ + c];
        }
        acc = siluf_(acc);
        int h = c >> 7, hd = c & 127;
        vssd[(((long long)b * H_ + h) * L_ + l) * HD_ + hd] = acc;
    }
}

__global__ __launch_bounds__(256)
void extract_bc_kernel(const float* __restrict__ proj, float* __restrict__ Bp,
                       float* __restrict__ Cp)
{
    long long row = blockIdx.x;                      // b*L + l
    int b = (int)(row / L_), l = (int)(row % L_);
    const float* pr = proj + row * PD_ + 2 * DI_;
    for (int i = threadIdx.x; i < 2 * H_ * N_; i += 256) {
        float v = pr[i];
        int which = i / (H_ * N_);
        int hn = i % (H_ * N_);
        int h = hn / N_, n = hn % N_;
        float* dst = which ? Cp : Bp;
        dst[(((long long)b * H_ + h) * L_ + l) * N_ + n] = v;
    }
}

__global__ void ld_kernel(const float* __restrict__ proj, const float* __restrict__ A_log,
                          const float* __restrict__ dt_bias, float* __restrict__ ld)
{
    long long idx = (long long)blockIdx.x * blockDim.x + threadIdx.x;  // B*L*H
    if (idx >= (long long)B_ * L_ * H_) return;
    int h = (int)(idx % H_);
    long long row = idx / H_;
    float p = proj[row * PD_ + 2 * DI_ + 2 * H_ * N_ + h] + dt_bias[h];
    float dt = softplusf_(p);
    int b = (int)(row / L_), l = (int)(row % L_);
    ld[((long long)b * H_ + h) * L_ + l] = -__expf(A_log[h]) * dt;
}

__global__ __launch_bounds__(256)
void cumsum_kernel(const float* __restrict__ ld, float* __restrict__ cum)
{
    __shared__ float buf[CS_];
    long long bh = blockIdx.x / NC_;
    int c = blockIdx.x % NC_;
    long long base = bh * L_ + (long long)c * CS_;
    int t = threadIdx.x;
    buf[t] = ld[base + t];
    __syncthreads();
    for (int o = 1; o < CS_; o <<= 1) {
        float add = (t >= o) ? buf[t - o] : 0.f;
        __syncthreads();
        buf[t] += add;
        __syncthreads();
    }
    cum[base + t] = buf[t];
}

__global__ void ssd_mask_kernel(float* __restrict__ Mm, const float* __restrict__ cum, int c)
{
    long long idx = (long long)blockIdx.x * 256 + threadIdx.x;   // B*H*CS*CS
    long long bh = idx >> 16;
    int tm = (int)(idx & 65535);
    int t = tm >> 8, m = tm & 255;
    const float* cc = cum + bh * L_ + (long long)c * CS_;
    float v = Mm[idx];
    Mm[idx] = (m <= t) ? v * __expf(cc[t] - cc[m]) : 0.f;
}

__global__ void ssd_vw_kernel(const float* __restrict__ vssd, const float* __restrict__ cum,
                              float* __restrict__ vw, int c)
{
    long long idx = (long long)blockIdx.x * 256 + threadIdx.x;   // B*H*CS*HD
    long long bh = idx / (CS_ * HD_);
    int r = (int)(idx % (CS_ * HD_));
    int l = r / HD_, d = r % HD_;
    const float* cc = cum + bh * L_ + (long long)c * CS_;
    float last = cc[CS_ - 1];
    vw[idx] = vssd[(bh * L_ + (long long)c * CS_ + l) * HD_ + d] * __expf(last - cc[l]);
}

__global__ void ssd_hscale_kernel(float* __restrict__ h, const float* __restrict__ cum, int c)
{
    long long idx = (long long)blockIdx.x * 256 + threadIdx.x;   // B*H*HD*N
    long long bh = idx / (HD_ * N_);
    float last = cum[bh * L_ + (long long)c * CS_ + CS_ - 1];
    h[idx] *= __expf(last);
}

__global__ void yg_kernel(const float* __restrict__ Y, const float* __restrict__ vssd,
                          const float* __restrict__ proj, const float* __restrict__ Dskip,
                          float* __restrict__ Yg)
{
    long long idx = (long long)blockIdx.x * 256 + threadIdx.x;   // B*L*DI
    long long row = idx / DI_;
    int di = (int)(idx % DI_);
    int h = di >> 7, hd = di & 127;
    int b = (int)(row / L_), l = (int)(row % L_);
    long long src = (((long long)b * H_ + h) * L_ + l) * HD_ + hd;
    float y = Y[src] + Dskip[di] * vssd[src];
    float gte = proj[row * PD_ + DI_ + di];
    Yg[idx] = y * siluf_(gte);
}

__global__ void wk_kernel(const float* __restrict__ out, float* __restrict__ wk)
{
    long long idx = (long long)blockIdx.x * 256 + threadIdx.x;   // B*L*D
    long long row = idx / D_;
    int l = (int)(row % L_);
    wk[idx] = (l == 0) ? 0.f : out[idx - D_];
}

__global__ void gplog_kernel(const float* __restrict__ decay, float* __restrict__ gplog)
{
    int t = threadIdx.x;                       // 0..255
    float lg = logf(sigmoidf_(decay[0]));
    gplog[t] = (float)t * lg;
}

__global__ void mem_masks_kernel(float* __restrict__ S, const float* __restrict__ decay)
{
    long long idx = (long long)blockIdx.x * 256 + threadIdx.x;   // B*MC*MC
    int ij = (int)(idx % (MC_ * MC_));
    int i = ij >> 8, j = ij & 255;
    float lg = logf(sigmoidf_(decay[0]));
    S[idx] = (i > j) ? S[idx] * __expf((float)(i - 1 - j) * lg) : 0.f;
}

__global__ void mem_vgw_kernel(const float* __restrict__ vwr, const float* __restrict__ decay,
                               float* __restrict__ vgw, int c)
{
    long long idx = (long long)blockIdx.x * 256 + threadIdx.x;   // B*MC*D
    long long b = idx / ((long long)MC_ * D_);
    int r = (int)(idx % ((long long)MC_ * D_));
    int t = r / D_, d = r % D_;
    float lg = logf(sigmoidf_(decay[0]));
    vgw[idx] = vwr[(b * L_ + (long long)c * MC_ + t) * D_ + d] *
               __expf((float)(MC_ - 1 - t) * lg);
}

__global__ void mem_scalew_kernel(float* __restrict__ W, const float* __restrict__ decay)
{
    long long idx = (long long)blockIdx.x * 256 + threadIdx.x;   // B*D*D
    float lg = logf(sigmoidf_(decay[0]));
    W[idx] *= __expf((float)MC_ * lg);
}

__global__ void final_kernel(const float* __restrict__ x, const float* __restrict__ out,
                             const float* __restrict__ rproj, float* __restrict__ o)
{
    long long idx = (long long)blockIdx.x * 256 + threadIdx.x;   // B*L*D
    o[idx] = x[idx] + out[idx] + ALPHA_ * rproj[idx];
}

// ---------------- host orchestration ----------------

static inline dim3 gemm_grid(int M, int N, int batches)
{
    return dim3((unsigned)((N + 63) / 64), (unsigned)((M + 63) / 64), (unsigned)batches);
}

extern "C" void kernel_launch(void* const* d_in, const int* in_sizes, int n_in,
                              void* d_out, int out_size, void* d_ws, size_t ws_size,
                              hipStream_t stream)
{
    (void)in_sizes; (void)n_in; (void)out_size; (void)ws_size;
    const float* x       = (const float*)d_in[0];
    const float* norm_w  = (const float*)d_in[1];
    const float* w_in    = (const float*)d_in[2];
    const float* conv_w  = (const float*)d_in[3];
    const float* conv_b  = (const float*)d_in[4];
    const float* A_log   = (const float*)d_in[5];
    const float* dt_bias = (const float*)d_in[6];
    const float* D_skip  = (const float*)d_in[7];
    const float* w_out   = (const float*)d_in[8];
    const float* w_write = (const float*)d_in[9];
    const float* w_read  = (const float*)d_in[10];
    const float* decay   = (const float*)d_in[11];
    float* outp = (float*)d_out;

    char* ws = (char*)d_ws;
    size_t off = 0;
    auto alloc = [&](size_t bytes) -> void* {
        void* p = ws + off;
        off = (off + bytes + 255) & ~(size_t)255;
        return p;
    };

    const long long BL = (long long)B_ * L_;

    // persistent (early+scan) buffers
    float* xn    = (float*)alloc((size_t)BL * D_ * 4);
    float* proj  = (float*)alloc((size_t)BL * PD_ * 4);
    float* vssd  = (float*)alloc((size_t)B_ * H_ * L_ * HD_ * 4);
    float* Bp    = (float*)alloc((size_t)B_ * H_ * L_ * N_ * 4);
    float* Cp    = (float*)alloc((size_t)B_ * H_ * L_ * N_ * 4);
    float* ldb   = (float*)alloc((size_t)B_ * H_ * L_ * 4);
    float* cum   = (float*)alloc((size_t)B_ * H_ * L_ * 4);
    float* Ybuf  = (float*)alloc((size_t)B_ * H_ * L_ * HD_ * 4);
    float* hst   = (float*)alloc((size_t)B_ * H_ * HD_ * N_ * 4);
    float* Mssd  = (float*)alloc((size_t)B_ * H_ * CS_ * CS_ * 4);
    float* vw    = (float*)alloc((size_t)B_ * H_ * CS_ * HD_ * 4);
    float* Yg    = (float*)alloc((size_t)BL * DI_ * 4);

    // late-phase buffers alias the [xn..proj] region (dead after yg_kernel)
    size_t loff = 0;
    auto lalloc = [&](size_t bytes) -> void* {
        void* p = ws + loff;
        loff = (loff + bytes + 255) & ~(size_t)255;
        return p;
    };
    float* outb  = (float*)lalloc((size_t)BL * D_ * 4);
    float* wkb   = (float*)lalloc((size_t)BL * D_ * 4);
    float* vwr   = (float*)lalloc((size_t)BL * D_ * 4);
    float* Wst   = (float*)lalloc((size_t)B_ * D_ * D_ * 4);
    float* Sbuf  = (float*)lalloc((size_t)B_ * MC_ * MC_ * 4);
    float* vgw   = (float*)lalloc((size_t)B_ * MC_ * D_ * 4);
    float* reads = (float*)lalloc((size_t)BL * D_ * 4);
    float* rproj = (float*)lalloc((size_t)BL * D_ * 4);
    float* gplog = (float*)lalloc(256 * 4);

    // ---- 1. rmsnorm + big input projection ----
    rmsnorm_kernel<<<(unsigned)BL, 256, 0, stream>>>(x, norm_w, xn);
    gemm_wmma_bf16<false, false><<<gemm_grid((int)BL, PD_, 1), 128, 0, stream>>>(
        xn, w_in, proj, (int)BL, PD_, D_, D_, D_, PD_, 0, 0, 0, 0.f, nullptr, 0);

    // ---- 2. conv+silu, B/C extraction, log-decays, per-chunk cumsum ----
    conv_silu_kernel<<<(unsigned)BL, 256, 0, stream>>>(proj, conv_w, conv_b, vssd);
    extract_bc_kernel<<<(unsigned)BL, 256, 0, stream>>>(proj, Bp, Cp);
    {
        long long n = (long long)B_ * L_ * H_;
        ld_kernel<<<(unsigned)((n + 255) / 256), 256, 0, stream>>>(proj, A_log, dt_bias, ldb);
    }
    cumsum_kernel<<<B_ * H_ * NC_, 256, 0, stream>>>(ldb, cum);

    // ---- 3. SSD chunked scan (batched WMMA GEMMs over B*H heads) ----
    hipMemsetAsync(hst, 0, (size_t)B_ * H_ * HD_ * N_ * 4, stream);
    for (int c = 0; c < NC_; ++c) {
        // Mraw = C_c @ B_c^T   (NT, batched over 32 heads)
        gemm_wmma_bf16<false, false><<<gemm_grid(CS_, CS_, B_ * H_), 128, 0, stream>>>(
            Cp + (long long)c * CS_ * N_, Bp + (long long)c * CS_ * N_, Mssd,
            CS_, CS_, N_, N_, N_, CS_,
            (long long)L_ * N_, (long long)L_ * N_, (long long)CS_ * CS_,
            0.f, nullptr, 0);
        ssd_mask_kernel<<<(B_ * H_ * CS_ * CS_) / 256, 256, 0, stream>>>(Mssd, cum, c);
        // Y_c = Mmasked @ v_c  (NN)
        gemm_wmma_bf16<false, true><<<gemm_grid(CS_, HD_, B_ * H_), 128, 0, stream>>>(
            Mssd, vssd + (long long)c * CS_ * HD_, Ybuf + (long long)c * CS_ * HD_,
            CS_, HD_, CS_, CS_, HD_, HD_,
            (long long)CS_ * CS_, (long long)L_ * HD_, (long long)L_ * HD_,
            0.f, nullptr, 0);
        // Y_c += exp(cum) * (C_c @ h^T)  (NT, per-row exp epilogue)
        gemm_wmma_bf16<false, false><<<gemm_grid(CS_, HD_, B_ * H_), 128, 0, stream>>>(
            Cp + (long long)c * CS_ * N_, hst, Ybuf + (long long)c * CS_ * HD_,
            CS_, HD_, N_, N_, N_, HD_,
            (long long)L_ * N_, (long long)HD_ * N_, (long long)L_ * HD_,
            1.f, cum + (long long)c * CS_, (long long)L_);
        // state: h = exp(last)*h + (v*exp(last-cum))^T @ B_c   (TN)
        {
            long long n = (long long)B_ * H_ * CS_ * HD_;
            ssd_vw_kernel<<<(unsigned)((n + 255) / 256), 256, 0, stream>>>(vssd, cum, vw, c);
            n = (long long)B_ * H_ * HD_ * N_;
            ssd_hscale_kernel<<<(unsigned)((n + 255) / 256), 256, 0, stream>>>(hst, cum, c);
        }
        gemm_wmma_bf16<true, true><<<gemm_grid(HD_, N_, B_ * H_), 128, 0, stream>>>(
            vw, Bp + (long long)c * CS_ * N_, hst,
            HD_, N_, CS_, HD_, N_, N_,
            (long long)CS_ * HD_, (long long)L_ * N_, (long long)HD_ * N_,
            1.f, nullptr, 0);
    }

    // ---- 4. gate + output projection chain ----
    {
        long long n = (long long)BL * DI_;
        yg_kernel<<<(unsigned)((n + 255) / 256), 256, 0, stream>>>(Ybuf, vssd, proj, D_skip, Yg);
    }
    gemm_wmma_bf16<false, false><<<gemm_grid((int)BL, D_, 1), 128, 0, stream>>>(
        Yg, w_out, outb, (int)BL, D_, DI_, DI_, DI_, D_, 0, 0, 0, 0.f, nullptr, 0);
    gemm_wmma_bf16<false, false><<<gemm_grid((int)BL, D_, 1), 128, 0, stream>>>(
        outb, w_write, vwr, (int)BL, D_, D_, D_, D_, D_, 0, 0, 0, 0.f, nullptr, 0);
    {
        long long n = (long long)BL * D_;
        wk_kernel<<<(unsigned)((n + 255) / 256), 256, 0, stream>>>(outb, wkb);
    }
    gplog_kernel<<<1, 256, 0, stream>>>(decay, gplog);

    // ---- 5. memory chunked scan (batched over B) ----
    hipMemsetAsync(Wst, 0, (size_t)B_ * D_ * D_ * 4, stream);
    for (int c = 0; c < NC_; ++c) {
        // inter = gamma^t * (rk_c @ W^T)
        gemm_wmma_bf16<false, false><<<gemm_grid(MC_, D_, B_), 128, 0, stream>>>(
            outb + (long long)c * MC_ * D_, Wst, reads + (long long)c * MC_ * D_,
            MC_, D_, D_, D_, D_, D_,
            (long long)L_ * D_, (long long)D_ * D_, (long long)L_ * D_,
            0.f, gplog, 0);
        // S = rk_c @ wk_c^T
        gemm_wmma_bf16<false, false><<<gemm_grid(MC_, MC_, B_), 128, 0, stream>>>(
            outb + (long long)c * MC_ * D_, wkb + (long long)c * MC_ * D_, Sbuf,
            MC_, MC_, D_, D_, D_, MC_,
            (long long)L_ * D_, (long long)L_ * D_, (long long)MC_ * MC_,
            0.f, nullptr, 0);
        {
            long long n = (long long)B_ * MC_ * MC_;
            mem_masks_kernel<<<(unsigned)((n + 255) / 256), 256, 0, stream>>>(Sbuf, decay);
        }
        // intra: reads_c += Smasked @ v_c  (NN)
        gemm_wmma_bf16<false, true><<<gemm_grid(MC_, D_, B_), 128, 0, stream>>>(
            Sbuf, vwr + (long long)c * MC_ * D_, reads + (long long)c * MC_ * D_,
            MC_, D_, MC_, MC_, D_, D_,
            (long long)MC_ * MC_, (long long)L_ * D_, (long long)L_ * D_,
            1.f, nullptr, 0);
        // W = gamma^MC * W + (v*gw)^T @ wk_c   (TN)
        {
            long long n = (long long)B_ * MC_ * D_;
            mem_vgw_kernel<<<(unsigned)((n + 255) / 256), 256, 0, stream>>>(vwr, decay, vgw, c);
            n = (long long)B_ * D_ * D_;
            mem_scalew_kernel<<<(unsigned)((n + 255) / 256), 256, 0, stream>>>(Wst, decay);
        }
        gemm_wmma_bf16<true, true><<<gemm_grid(D_, D_, B_), 128, 0, stream>>>(
            vgw, wkb + (long long)c * MC_ * D_, Wst,
            D_, D_, MC_, D_, D_, D_,
            (long long)MC_ * D_, (long long)L_ * D_, (long long)D_ * D_,
            1.f, nullptr, 0);
    }

    // ---- 6. read projection + residual ----
    gemm_wmma_bf16<false, false><<<gemm_grid((int)BL, D_, 1), 128, 0, stream>>>(
        reads, w_read, rproj, (int)BL, D_, D_, D_, D_, D_, 0, 0, 0, 0.f, nullptr, 0);
    {
        long long n = (long long)BL * D_;
        final_kernel<<<(unsigned)((n + 255) / 256), 256, 0, stream>>>(x, outb, rproj, outp);
    }
}
